// AttnReadout_72129680769667
// MI455X (gfx1250) — matrix operand order repaired and checked by reading, not compiled, not run on hardware
//
#include <hip/hip_runtime.h>
#include <stdint.h>

#define B_ 256
#define N_ 512
#define D_ 256
#define CHUNK 32                    // rows of x staged per LDS pass
#define NCHUNK (N_ / CHUNK)         // 16

typedef __attribute__((ext_vector_type(16))) __bf16 v16bf;
typedef __attribute__((ext_vector_type(8)))  float  v8f;

struct alignas(32) Frag32 { uint4 lo, hi; };

// f32 -> bf16 round-to-nearest-even
__device__ __forceinline__ unsigned short f2bf(float f) {
    unsigned int u = __builtin_bit_cast(unsigned int, f);
    u += 0x7FFFu + ((u >> 16) & 1u);
    return (unsigned short)(u >> 16);
}

// fast sigmoid: v_exp_f32 + v_rcp_f32 (avoid IEEE divide expansion)
__device__ __forceinline__ float fast_sigmoid(float z) {
    return __builtin_amdgcn_rcpf(1.0f + __expf(-z));
}

// W_u (f32, [K=256][Dout=256] row-major) -> bf16 transposed [Dout][K]
// so each B-fragment half is one contiguous 16B run along K.
__global__ void prep_weights(const float* __restrict__ Wu,
                             unsigned short* __restrict__ WuT) {
    const int dout = blockIdx.x;
    const int k    = threadIdx.x;
    WuT[dout * D_ + k] = f2bf(Wu[k * D_ + dout]);
}

// x_v[b][d] = x[b, last[b], :] @ W_v[:, d]
__global__ void compute_xv(const float* __restrict__ x, const int* __restrict__ last,
                           const float* __restrict__ Wv, float* __restrict__ xv) {
    const int b = blockIdx.x;
    const int d = threadIdx.x;
    const float* row = x + ((long)b * N_ + (long)last[b]) * D_;
    float acc = 0.f;
    #pragma unroll 8
    for (int k = 0; k < D_; ++k) acc += row[k] * Wv[k * D_ + d];
    xv[b * D_ + d] = acc;
}

// One workgroup per batch. 8 waves (wave32).
__global__ void __launch_bounds__(256)
attn_readout(const float* __restrict__ x, const float* __restrict__ bu,
             const float* __restrict__ We, const unsigned short* __restrict__ WuT,
             const float* __restrict__ xv, float* __restrict__ out) {
    __shared__ unsigned short wub[D_ * D_];     // 128 KB bf16 W_u^T (whole matrix)
    __shared__ unsigned short xs[CHUNK * D_];   //  16 KB bf16 x tile
    __shared__ float e[N_];                     // logits -> alpha (2 KB)
    __shared__ float red[256];                  // reduction scratch (1 KB)

    const int  b    = blockIdx.x;
    const int  t    = threadIdx.x;
    const int  wave = t >> 5;
    const int  lane = t & 31;
    const int  lo   = lane & 15;   // column lane (N of C tile / col of B)
    const int  hi   = lane >> 4;   // lane-half -> K split in A/B fragments
    const long xbase = (long)b * N_ * D_;

    // ---- async DMA of bf16 W_u^T into LDS (gfx1250 async-to-LDS path) ----
    // 128 KB = 8192 x b128; 32 instructions per thread, tracked by ASYNCcnt.
    {
        const uint4* wsrc = (const uint4*)WuT;
        #pragma unroll
        for (int i = 0; i < (D_ * D_ * 2 / 16) / 256; ++i) {   // 32 iters
            const int idx = i * 256 + t;
            unsigned long long ga = (unsigned long long)(wsrc + idx);
            unsigned ldsoff = (unsigned)(uintptr_t)(((uint4*)wub) + idx);
            asm volatile("global_load_async_to_lds_b128 %0, %1, off"
                         :: "v"(ldsoff), "v"(ga) : "memory");
        }
        asm volatile("s_wait_asynccnt 0x0" ::: "memory");
    }

    e[t] = 0.f;
    e[t + 256] = 0.f;

    const int tile = wave & 1;              // n-tile (16 rows) within chunk
    const int dt0  = (wave >> 1) * 4;       // this wave covers 4 of 16 d-tiles

    for (int chunk = 0; chunk < NCHUNK; ++chunk) {
        __syncthreads();                    // LDS tile safe to overwrite
        // ---- stage 32x256 fp32 -> bf16 LDS (coalesced float4 reads) ----
        const float4* src = (const float4*)(x + xbase + (long)chunk * CHUNK * D_);
        #pragma unroll
        for (int i = 0; i < (CHUNK * D_) / (256 * 4); ++i) {   // 8 iters
            const int v = i * 256 + t;
            float4 f = src[v];
            ushort4 h;
            h.x = f2bf(f.x); h.y = f2bf(f.y); h.z = f2bf(f.z); h.w = f2bf(f.w);
            ((ushort4*)xs)[v] = h;
        }
        __syncthreads();

        float partial[8];
        #pragma unroll
        for (int r = 0; r < 8; ++r) partial[r] = 0.f;

        const int arow = tile * 16 + lo;    // A-matrix row M for this lane

        for (int dti = 0; dti < 4; ++dti) {
            const int dtile = dt0 + dti;
            v8f c = {0.f, 0.f, 0.f, 0.f, 0.f, 0.f, 0.f, 0.f};
            #pragma unroll
            for (int kt = 0; kt < 8; ++kt) {           // K = 256 in steps of 32
                // A fragment (ISA 16-bit A layout): lane(lo)=M,
                // K chunks [8*hi, 8*hi+8) and [16+8*hi, 16+8*hi+8)
                const unsigned short* arp = xs + arow * D_ + kt * 32;
                Frag32 fa;
                fa.lo = *(const uint4*)(arp + 8 * hi);
                fa.hi = *(const uint4*)(arp + 16 + 8 * hi);
                // B fragment from LDS: lane(lo)=col, K chunks [8*hi, 8*hi+8) and [16+8*hi, 16+8*hi+8)
                const unsigned short* brp = wub + (dtile * 16 + lo) * D_ + kt * 32 + 8 * hi;
                Frag32 fb;
                fb.lo = *(const uint4*)(brp);
                fb.hi = *(const uint4*)(brp + 16);
                v16bf a  = __builtin_bit_cast(v16bf, fa);
                v16bf bm = __builtin_bit_cast(v16bf, fb);
                c = __builtin_amdgcn_wmma_f32_16x16x32_bf16(
                        false, a, false, bm, (short)0, c, false, false);
            }
            // epilogue: z = xu + b_u[d] + xv[b][d]; partial += sigmoid(z)*W_e[d]
            const int d = dtile * 16 + lo;
            const float addv = bu[d] + xv[b * D_ + d];
            const float w    = We[d];
            #pragma unroll
            for (int r = 0; r < 8; ++r) {              // row m = r + 8*hi
                partial[r] += fast_sigmoid(c[r] + addv) * w;
            }
        }
        // reduce the 16 column-lanes -> per-row logit contribution
        #pragma unroll
        for (int r = 0; r < 8; ++r) {
            float v = partial[r];
            v += __shfl_xor(v, 1);
            v += __shfl_xor(v, 2);
            v += __shfl_xor(v, 4);
            v += __shfl_xor(v, 8);
            if (lo == 0)
                atomicAdd(&e[chunk * CHUNK + tile * 16 + r + 8 * hi], v);
        }
    }
    __syncthreads();

    // ---- softmax over N=512 (each thread owns e[t], e[t+256]) ----
    float m = fmaxf(e[t], e[t + 256]);
    red[t] = m;
    __syncthreads();
    for (int s = 128; s > 0; s >>= 1) {
        if (t < s) red[t] = fmaxf(red[t], red[t + s]);
        __syncthreads();
    }
    const float M = red[0];
    __syncthreads();
    const float a0 = __expf(e[t] - M);
    const float a1 = __expf(e[t + 256] - M);
    red[t] = a0 + a1;
    __syncthreads();
    for (int s = 128; s > 0; s >>= 1) {
        if (t < s) red[t] += red[t + s];
        __syncthreads();
    }
    const float inv = __builtin_amdgcn_rcpf(red[0]);
    __syncthreads();
    e[t]       = a0 * inv;
    e[t + 256] = a1 * inv;
    __syncthreads();

    // ---- out[b][d] = sum_n alpha[n] * x[b][n][d]  (coalesced, L2-resident) ----
    float acc = 0.f;
    const float* xp = x + xbase + t;
    #pragma unroll 4
    for (int n = 0; n < N_; ++n) acc += e[n] * xp[(long)n * D_];
    out[b * D_ + t] = acc;
}

extern "C" void kernel_launch(void* const* d_in, const int* in_sizes, int n_in,
                              void* d_out, int out_size, void* d_ws, size_t ws_size,
                              hipStream_t stream) {
    (void)in_sizes; (void)n_in; (void)out_size; (void)ws_size;
    const float* x    = (const float*)d_in[0];
    const int*   last = (const int*)  d_in[1];
    const float* Wu   = (const float*)d_in[2];
    const float* bu   = (const float*)d_in[3];
    const float* Wv   = (const float*)d_in[4];
    const float* We   = (const float*)d_in[5];
    float* out = (float*)d_out;

    char* ws = (char*)d_ws;
    unsigned short* WuT = (unsigned short*)ws;                                     // 128 KB
    float*          xv  = (float*)(ws + (size_t)D_ * D_ * sizeof(unsigned short)); // 256 KB

    prep_weights<<<D_, D_, 0, stream>>>(Wu, WuT);
    compute_xv<<<B_, D_, 0, stream>>>(x, last, Wv, xv);
    attn_readout<<<B_, 256, 0, stream>>>(x, bu, We, WuT, xv, out);
}
